// EMMA_47072841564938
// MI455X (gfx1250) — matrix operand order repaired and checked by reading, not compiled, not run on hardware
//
#include <hip/hip_runtime.h>
#include <hip/hip_bf16.h>
#include <math.h>

// ---------------------------------------------------------------------------
// Problem constants (from the reference)
// ---------------------------------------------------------------------------
#define BB     64
#define LL     128
#define EMB    512
#define HID    1024
#define MEMD   512
#define NVAL   1024
#define NSLOTS 256
#define KTOP   16
#define HDN    2048
#define DEQ_ITERS 15
#define RELAX  0.5f
#define DECAY  0.995f

typedef __attribute__((ext_vector_type(16))) __bf16 v16bf;
typedef __attribute__((ext_vector_type(8)))  float  v8f;

__device__ __forceinline__ unsigned short f2bf(float f) {
    unsigned int u = __float_as_uint(f);
    unsigned int r = u + 0x7fffu + ((u >> 16) & 1u);   // round-to-nearest-even
    return (unsigned short)(r >> 16);
}

// ---------------------------------------------------------------------------
// Weight repack (one-time per call): W f32 [K][N] row-major -> bf16 tiles in
// a WMMA B-fragment layout (16-bit B 32x16, contiguous k order, same as pack_a):
//   lanes 0-15:  n = lane,     element e -> k = kt*32 + e
//   lanes 16-31: n = lane-16,  element e -> k = kt*32 + 16 + e
// Tile (kt,nt) stored as 32 lanes x 16 bf16 contiguous (32 B/lane, coalesced).
// ---------------------------------------------------------------------------
__global__ void __launch_bounds__(256) repack_b(const float* __restrict__ W,
                                                unsigned short* __restrict__ Wp,
                                                int K, int N) {
    int tile = blockIdx.x * 8 + (threadIdx.x >> 5);
    int lane = threadIdx.x & 31;
    int ntiles = N >> 4;
    int kt = tile / ntiles, nt = tile % ntiles;
    if (kt >= (K >> 5)) return;
    int n  = nt * 16 + (lane & 15);
    int hi = lane >> 4;
    unsigned short* dst = Wp + ((size_t)tile * 32 + lane) * 16;
#pragma unroll
    for (int e = 0; e < 16; ++e) {
        int k = kt * 32 + 16 * hi + e;
        dst[e] = f2bf(W[(size_t)k * N + n]);
    }
}

// ---------------------------------------------------------------------------
// Activation pack (per GEMM): A f32 [64,K] -> bf16 tiles in the WMMA
// A-fragment layout (16-bit A 16x32, contiguous k order, same as repack_b):
//   lane = (m & 15) + 16*hi ; element e -> k = 16*hi+e
// Tile index = mt*(K/32) + kt (K-contiguous per M-tile for streaming).
// Loads: 4x float4 per lane; store: 2x b128 per lane (coalesced).
// ---------------------------------------------------------------------------
__global__ void __launch_bounds__(256) pack_a(const float* __restrict__ A, int K,
                                              unsigned short* __restrict__ Ap) {
    int tile = blockIdx.x * 8 + (threadIdx.x >> 5);
    int lane = threadIdx.x & 31;
    int nkt  = K >> 5;
    int mt = tile / nkt, kt = tile % nkt;
    if (mt >= (BB / 16)) return;
    int m  = mt * 16 + (lane & 15);
    int hi = lane >> 4;
    const float* src = A + (size_t)m * K + kt * 32;
    const float4* s0 = (const float4*)(src + 16 * hi);       // k = 16hi .. 16hi+7
    const float4* s1 = (const float4*)(src + 16 * hi + 8);   // k = 16hi+8 ..
    float4 a0 = s0[0], a1 = s0[1], b0 = s1[0], b1 = s1[1];
    union { uint4 q[2]; unsigned short u[16]; } o;
    o.u[0]  = f2bf(a0.x); o.u[1]  = f2bf(a0.y); o.u[2]  = f2bf(a0.z); o.u[3]  = f2bf(a0.w);
    o.u[4]  = f2bf(a1.x); o.u[5]  = f2bf(a1.y); o.u[6]  = f2bf(a1.z); o.u[7]  = f2bf(a1.w);
    o.u[8]  = f2bf(b0.x); o.u[9]  = f2bf(b0.y); o.u[10] = f2bf(b0.z); o.u[11] = f2bf(b0.w);
    o.u[12] = f2bf(b1.x); o.u[13] = f2bf(b1.y); o.u[14] = f2bf(b1.z); o.u[15] = f2bf(b1.w);
    uint4* dst = (uint4*)(Ap + ((size_t)tile * 32 + lane) * 16);
    dst[0] = o.q[0];  dst[1] = o.q[1];
}

// ---------------------------------------------------------------------------
// Generic fused WMMA GEMM: C = act( alpha*(A@Wp + bias + addend) + c0w*C0 )
// Ap: packed bf16 A-frags.  Wp: packed bf16 B-frags.  C: [64,N] f32.
// Block = 256 thr = 8 waves; each wave -> one 16x32 C strip (2 N-tiles,
// A-frag reused by both WMMAs).  grid = (N/256, 4).  No LDS, no barriers:
// inner loop = 3x b128 loads per v_wmma_f32_16x16x32_bf16.
// act: 0=none 1=tanh 2=sigmoid.
// ---------------------------------------------------------------------------
__global__ void __launch_bounds__(256) wmma_gemm(
        const unsigned short* __restrict__ Ap, int K,
        const unsigned short* __restrict__ Wp, int N,
        const float* __restrict__ bias,
        const float* __restrict__ addend,
        const float* __restrict__ C0, float c0w,
        float alpha, int act,
        float* __restrict__ C) {
    const int wave = threadIdx.x >> 5;
    const int lane = threadIdx.x & 31;
    const int mt   = blockIdx.y;
    const int nt0  = (blockIdx.x * 8 + wave) * 2;
    const int ntiles = N >> 4;
    const int nkt    = K >> 5;

    v8f acc0 = {}, acc1 = {};
    const uint4* ap = (const uint4*)(Ap + (((size_t)mt * nkt) * 32 + lane) * 16);
    const uint4* bp = (const uint4*)(Wp + ((size_t)nt0 * 32 + lane) * 16);
    const size_t astep = 64;                       // one ktile = 1024 B = 64 uint4
    const size_t bstep = (size_t)ntiles * 64;      // advance one kt row of tiles

    for (int kt = 0; kt < nkt; ++kt) {
        union { v16bf v; uint4 q[2]; } a, b0, b1;
        a.q[0]  = ap[0];            a.q[1]  = ap[1];
        b0.q[0] = bp[0];            b0.q[1] = bp[1];
        b1.q[0] = bp[32];           b1.q[1] = bp[33];   // next N-tile (+512 B)
        acc0 = __builtin_amdgcn_wmma_f32_16x16x32_bf16(
                   false, a.v, false, b0.v, (short)0, acc0, false, false);
        acc1 = __builtin_amdgcn_wmma_f32_16x16x32_bf16(
                   false, a.v, false, b1.v, (short)0, acc1, false, false);
        ap += astep;  bp += bstep;
    }
    // epilogue: C/D layout -> VGPR r holds row mt*16 + r + 8*hi, col = lane&15
    const int hi = lane >> 4;
    const int rbase = mt * 16 + 8 * hi;
#pragma unroll
    for (int tile2 = 0; tile2 < 2; ++tile2) {
        const int n = (nt0 + tile2) * 16 + (lane & 15);
        const float bn = bias ? bias[n] : 0.f;
        v8f acc = tile2 ? acc1 : acc0;
#pragma unroll
        for (int r = 0; r < 8; ++r) {
            int row = rbase + r;
            float v = acc[r] + bn;
            if (addend) v += addend[(size_t)row * N + n];
            v *= alpha;
            if (C0) v += c0w * C0[(size_t)row * N + n];
            if (act == 1)      v = tanhf(v);
            else if (act == 2) v = 1.f / (1.f + __expf(-v));
            C[(size_t)row * N + n] = v;
        }
    }
}

// ---------------------------------------------------------------------------
// Setup kernels
// ---------------------------------------------------------------------------
__global__ void norm_rows(const float* __restrict__ in, float* __restrict__ out, int D) {
    int r = blockIdx.x, tid = threadIdx.x;            // block = 128
    __shared__ float red[128];
    float ss = 0.f;
    for (int d = tid; d < D; d += 128) { float x = in[(size_t)r * D + d]; ss += x * x; }
    red[tid] = ss; __syncthreads();
    for (int s = 64; s > 0; s >>= 1) { if (tid < s) red[tid] += red[tid + s]; __syncthreads(); }
    float inv = 1.f / (sqrtf(red[0]) + 1e-12f);
    for (int d = tid; d < D; d += 128) out[(size_t)r * D + d] = in[(size_t)r * D + d] * inv;
}

// Per-sequence: kvec = norm(key_embed[key_id]); top-16 slots of kvec@sk^T; softmax.
__global__ void __launch_bounds__(256) prep_slots(
        const int* __restrict__ key_ids, const float* __restrict__ key_embed,
        const float* __restrict__ slot_keys,
        float* __restrict__ kvec, int* __restrict__ topi, float* __restrict__ w_slot) {
    int b = blockIdx.x, tid = threadIdx.x;
    __shared__ float kv[MEMD];
    __shared__ float red[256]; __shared__ int redi[256];
    __shared__ float sims[NSLOTS]; __shared__ float topv_s[KTOP];
    const float* krow = key_embed + (size_t)key_ids[b] * MEMD;
    float ss = 0.f;
    for (int d = tid; d < MEMD; d += 256) { float x = krow[d]; kv[d] = x; ss += x * x; }
    red[tid] = ss; __syncthreads();
    for (int s = 128; s > 0; s >>= 1) { if (tid < s) red[tid] += red[tid + s]; __syncthreads(); }
    float inv = 1.f / (sqrtf(red[0]) + 1e-12f);
    __syncthreads();
    for (int d = tid; d < MEMD; d += 256) { kv[d] *= inv; kvec[b * MEMD + d] = kv[d]; }
    __syncthreads();
    {   // one slot per thread
        const float* srow = slot_keys + (size_t)tid * MEMD;
        float s = 0.f;
        for (int d = 0; d < MEMD; ++d) s += kv[d] * srow[d];
        sims[tid] = s;
    }
    __syncthreads();
    for (int it = 0; it < KTOP; ++it) {
        red[tid] = sims[tid]; redi[tid] = tid; __syncthreads();
        for (int s = 128; s > 0; s >>= 1) {
            if (tid < s) {
                if (red[tid + s] > red[tid] ||
                    (red[tid + s] == red[tid] && redi[tid + s] < redi[tid])) {
                    red[tid] = red[tid + s]; redi[tid] = redi[tid + s];
                }
            }
            __syncthreads();
        }
        if (tid == 0) { topi[b * KTOP + it] = redi[0]; topv_s[it] = red[0]; sims[redi[0]] = -3.4e38f; }
        __syncthreads();
    }
    if (tid == 0) {
        float mx = topv_s[0], sum = 0.f, e_[KTOP];
        for (int k = 0; k < KTOP; ++k) { e_[k] = __expf(topv_s[k] - mx); sum += e_[k]; }
        for (int k = 0; k < KTOP; ++k) w_slot[b * KTOP + k] = e_[k] / sum;
    }
}

__global__ void calc_scale(const float* __restrict__ raw, float* __restrict__ scale) {
    if (threadIdx.x == 0) {
        float x = raw[0];
        float sp = (x > 20.f) ? x : log1pf(__expf(x));   // softplus
        scale[0] = sp + 1e-3f;
    }
}

__global__ void init_state(const float* __restrict__ h0, float* __restrict__ z,
                           float* __restrict__ h, float* __restrict__ memory,
                           float* __restrict__ out) {
    int i = blockIdx.x * blockDim.x + threadIdx.x;
    if (i < BB * HID) { z[i] = 0.f; h[i] = h0[i & (HID - 1)]; }
    if (i < BB * NVAL) out[i] = 0.f;
    if (i < NSLOTS * MEMD) memory[i] = 0.f;
}

// ---------------------------------------------------------------------------
// Per-step kernels
// ---------------------------------------------------------------------------
__global__ void gather_x(int t, const int* __restrict__ tokens,
                         const float* __restrict__ embed, float* __restrict__ xbuf) {
    int b = blockIdx.x;
    int tok = tokens[b * LL + t];
    for (int d = threadIdx.x; d < EMB; d += blockDim.x)
        xbuf[b * EMB + d] = embed[(size_t)tok * EMB + d];
}

// VSA read + cleanup (query rows only). argmax over v_raw@Vp^T equals argmax
// over norm(v_raw)@Vp^T (positive row scaling); Vp rows are unit -> inject=Vp[idx].
__global__ void __launch_bounds__(256) mem_read_k(
        int t, const int* __restrict__ query_pos, const int* __restrict__ topi,
        const float* __restrict__ w_slot, const float* __restrict__ memory,
        const float* __restrict__ kvec, const float* __restrict__ Vp,
        float* __restrict__ vbuf) {
    int b = blockIdx.x;
    if (query_pos[b] != t) return;
    int tid = threadIdx.x;
    __shared__ float vr[MEMD];
    __shared__ float red[256]; __shared__ int redi[256];
    for (int d = tid; d < MEMD; d += 256) {
        float m = 0.f;
        for (int k = 0; k < KTOP; ++k)
            m += w_slot[b * KTOP + k] * memory[(size_t)topi[b * KTOP + k] * MEMD + d];
        vr[d] = m * kvec[b * MEMD + d];
    }
    __syncthreads();
    float best = -3.4e38f; int bi = 0;
    for (int v = tid; v < NVAL; v += 256) {
        const float* pr = Vp + (size_t)v * MEMD;
        float s = 0.f;
        for (int d = 0; d < MEMD; ++d) s += vr[d] * pr[d];
        if (s > best) { best = s; bi = v; }
    }
    red[tid] = best; redi[tid] = bi; __syncthreads();
    for (int s = 128; s > 0; s >>= 1) {
        if (tid < s) {
            if (red[tid + s] > red[tid] ||
                (red[tid + s] == red[tid] && redi[tid + s] < redi[tid])) {
                red[tid] = red[tid + s]; redi[tid] = redi[tid + s];
            }
        }
        __syncthreads();
    }
    int idx = redi[0];
    for (int d = tid; d < MEMD; d += 256)
        vbuf[b * MEMD + d] = Vp[(size_t)idx * MEMD + d];   // MEM_SCALE = 1
}

// pre += v_t @ Wv  (~1 active row per step -> scalar path, Wv kept f32)
__global__ void add_v(int t, const int* __restrict__ query_pos,
                      const float* __restrict__ vbuf, const float* __restrict__ Wv,
                      float* __restrict__ pre) {
    int b = blockIdx.y;
    if (query_pos[b] != t) return;
    int j = blockIdx.x * blockDim.x + threadIdx.x;     // 0..HDN-1
    float s = 0.f;
    for (int m = 0; m < MEMD; ++m) s += vbuf[b * MEMD + m] * Wv[(size_t)m * HDN + j];
    pre[(size_t)b * HDN + j] += s;
}

__global__ void h_update(const float* __restrict__ g, const float* __restrict__ l,
                         float* __restrict__ h) {
    int i = blockIdx.x * blockDim.x + threadIdx.x;
    float gg = g[i];
    h[i] = (1.f - gg) * h[i] + gg * l[i];
}

__global__ void concat_zh(const float* __restrict__ z, const float* __restrict__ h,
                          float* __restrict__ zh) {
    int i = blockIdx.x * blockDim.x + threadIdx.x;     // BB*HDN
    int b = i >> 11, c = i & (HDN - 1);
    zh[i] = (c < HID) ? z[b * HID + c] : h[b * HID + (c - HID)];
}

__global__ void mark_written(int t, const int* __restrict__ write_pos,
                             const int* __restrict__ topi, float* __restrict__ written) {
    int tid = threadIdx.x;                              // single block of 256
    written[tid] = 0.f;
    __syncthreads();
    if (tid < BB && write_pos[tid] == t)
        for (int k = 0; k < KTOP; ++k) written[topi[tid * KTOP + k]] = 1.f;
}

__global__ void apply_decay(const float* __restrict__ written, float* __restrict__ memory) {
    int s = blockIdx.x, tid = threadIdx.x;              // 256 blocks x 256 thr
    float dec = 1.f - written[s] * (1.f - DECAY);
    memory[(size_t)s * MEMD + tid]       *= dec;
    memory[(size_t)s * MEMD + 256 + tid] *= dec;
}

// Deterministic scatter: each thread owns dim d, iterates sequences in fixed
// order (no float atomics -> bit-identical across graph replays).
__global__ void scatter_write(int t, const int* __restrict__ write_pos,
                              const int* __restrict__ topi, const float* __restrict__ w_slot,
                              const float* __restrict__ kvec, const float* __restrict__ vpred,
                              float* __restrict__ memory) {
    int d = blockIdx.x * blockDim.x + threadIdx.x;      // 0..MEMD-1
    for (int b = 0; b < BB; ++b) {
        if (write_pos[b] != t) continue;
        float bind = kvec[b * MEMD + d] * vpred[b * MEMD + d];
        for (int k = 0; k < KTOP; ++k) {
            int s = topi[b * KTOP + k];
            memory[(size_t)s * MEMD + d] += w_slot[b * KTOP + k] * bind;
        }
    }
}

__global__ void logits_k(int t, const int* __restrict__ query_pos,
                         const float* __restrict__ vpred, const float* __restrict__ Vp,
                         const float* __restrict__ scale, float* __restrict__ out) {
    int b = blockIdx.y;
    if (query_pos[b] != t) return;
    int v = blockIdx.x * blockDim.x + threadIdx.x;     // 0..NVAL-1
    const float* pr = Vp + (size_t)v * MEMD;
    float s = 0.f;
    for (int d = 0; d < MEMD; ++d) s += vpred[b * MEMD + d] * pr[d];
    out[b * NVAL + v] = scale[0] * s;
}

// ---------------------------------------------------------------------------
// Host: graph-capturable stream, ~80 kernels per time step.
// ---------------------------------------------------------------------------
extern "C" void kernel_launch(void* const* d_in, const int* in_sizes, int n_in,
                              void* d_out, int out_size, void* d_ws, size_t ws_size,
                              hipStream_t stream) {
    (void)in_sizes; (void)n_in; (void)out_size; (void)ws_size;
    const int*   tokens    = (const int*)d_in[0];
    const int*   key_ids   = (const int*)d_in[1];
    const int*   write_pos = (const int*)d_in[3];
    const int*   query_pos = (const int*)d_in[4];
    const float* embed     = (const float*)d_in[5];
    const float* key_embed = (const float*)d_in[6];
    const float* value_emb = (const float*)d_in[7];
    const float* Wz  = (const float*)d_in[8];
    const float* Wx  = (const float*)d_in[9];
    const float* Wv  = (const float*)d_in[10];
    const float* b1  = (const float*)d_in[11];
    const float* Wo  = (const float*)d_in[12];
    const float* bo  = (const float*)d_in[13];
    const float* Wg  = (const float*)d_in[14];
    const float* Ug  = (const float*)d_in[15];
    const float* bg  = (const float*)d_in[16];
    const float* Wl  = (const float*)d_in[17];
    const float* Ul  = (const float*)d_in[18];
    const float* bl  = (const float*)d_in[19];
    const float* Wzv = (const float*)d_in[20];
    const float* bzv = (const float*)d_in[21];
    const float* h0  = (const float*)d_in[22];
    const float* lsr = (const float*)d_in[23];
    const float* sk  = (const float*)d_in[24];
    float* out = (float*)d_out;

    char* ws = (char*)d_ws;
    size_t off = 0;
    auto alloc = [&](size_t bytes) -> void* {
        void* p = ws + off; off = (off + bytes + 255) & ~(size_t)255; return p;
    };
    // bf16-packed weights (~20 MB total; resident in the 192 MB L2 at run time)
    unsigned short* Wz_p  = (unsigned short*)alloc((size_t)HID * HDN * 2);
    unsigned short* Wx_p  = (unsigned short*)alloc((size_t)EMB * HDN * 2);
    unsigned short* Wo_p  = (unsigned short*)alloc((size_t)HDN * HID * 2);
    unsigned short* Wg_p  = (unsigned short*)alloc((size_t)HID * HID * 2);
    unsigned short* Ug_p  = (unsigned short*)alloc((size_t)HID * HID * 2);
    unsigned short* Wl_p  = (unsigned short*)alloc((size_t)HID * HID * 2);
    unsigned short* Ul_p  = (unsigned short*)alloc((size_t)HID * HID * 2);
    unsigned short* Wzv_p = (unsigned short*)alloc((size_t)HDN * MEMD * 2);
    // bf16-packed activation fragments
    unsigned short* xp  = (unsigned short*)alloc((size_t)BB * EMB * 2);
    unsigned short* zp  = (unsigned short*)alloc((size_t)BB * HID * 2);
    unsigned short* hp  = (unsigned short*)alloc((size_t)BB * HID * 2);
    unsigned short* dp  = (unsigned short*)alloc((size_t)BB * HDN * 2);   // hdn
    unsigned short* zhp = (unsigned short*)alloc((size_t)BB * HDN * 2);
    // f32 state / activations
    float* Vp     = (float*)alloc((size_t)NVAL * MEMD * 4);
    float* kvec   = (float*)alloc((size_t)BB * MEMD * 4);
    int*   topi   = (int*)  alloc((size_t)BB * KTOP * 4);
    float* w_slot = (float*)alloc((size_t)BB * KTOP * 4);
    float* memory = (float*)alloc((size_t)NSLOTS * MEMD * 4);
    float* z      = (float*)alloc((size_t)BB * HID * 4);
    float* h      = (float*)alloc((size_t)BB * HID * 4);
    float* hdn    = (float*)alloc((size_t)BB * HDN * 4);
    float* pre    = (float*)alloc((size_t)BB * HDN * 4);
    float* tg     = (float*)alloc((size_t)BB * HID * 4);
    float* gbuf   = (float*)alloc((size_t)BB * HID * 4);
    float* tl     = (float*)alloc((size_t)BB * HID * 4);
    float* lbuf   = (float*)alloc((size_t)BB * HID * 4);
    float* zh     = (float*)alloc((size_t)BB * HDN * 4);
    float* vtmp   = (float*)alloc((size_t)BB * MEMD * 4);
    float* vpred  = (float*)alloc((size_t)BB * MEMD * 4);
    float* xbuf   = (float*)alloc((size_t)BB * EMB * 4);
    float* vbuf   = (float*)alloc((size_t)BB * MEMD * 4);
    float* written= (float*)alloc((size_t)NSLOTS * 4);
    float* scale  = (float*)alloc(256);

    auto repack = [&](const float* W, unsigned short* Wp, int K, int N) {
        int tiles = (K / 32) * (N / 16);
        repack_b<<<(tiles + 7) / 8, 256, 0, stream>>>(W, Wp, K, N);
    };
    auto pack = [&](const float* A, int K, unsigned short* Ap) {
        int tiles = (BB / 16) * (K / 32);
        pack_a<<<(tiles + 7) / 8, 256, 0, stream>>>(A, K, Ap);
    };
    auto gemm = [&](const unsigned short* Ap, int K, const unsigned short* Wp, int N,
                    const float* bias, const float* addend, const float* C0,
                    float c0w, float alpha, int act, float* C) {
        wmma_gemm<<<dim3(N / 256, BB / 16), 256, 0, stream>>>(
            Ap, K, Wp, N, bias, addend, C0, c0w, alpha, act, C);
    };

    // ---- one-time setup (re-run every call: deterministic, no cached state)
    repack(Wz,  Wz_p,  HID, HDN);
    repack(Wx,  Wx_p,  EMB, HDN);
    repack(Wo,  Wo_p,  HDN, HID);
    repack(Wg,  Wg_p,  HID, HID);
    repack(Ug,  Ug_p,  HID, HID);
    repack(Wl,  Wl_p,  HID, HID);
    repack(Ul,  Ul_p,  HID, HID);
    repack(Wzv, Wzv_p, HDN, MEMD);
    norm_rows<<<NVAL, 128, 0, stream>>>(value_emb, Vp, MEMD);
    prep_slots<<<BB, 256, 0, stream>>>(key_ids, key_embed, sk, kvec, topi, w_slot);
    calc_scale<<<1, 32, 0, stream>>>(lsr, scale);
    init_state<<<(NSLOTS * MEMD) / 256, 256, 0, stream>>>(h0, z, h, memory, out);

    // ---- sequential scan over time steps
    for (int t = 0; t < LL; ++t) {
        gather_x<<<BB, 128, 0, stream>>>(t, tokens, embed, xbuf);
        mem_read_k<<<BB, 256, 0, stream>>>(t, query_pos, topi, w_slot, memory, kvec, Vp, vbuf);
        // pre = x_t @ Wx + b1  (+ v_t @ Wv for query rows)
        pack(xbuf, EMB, xp);
        gemm(xp, EMB, Wx_p, HDN, b1, nullptr, nullptr, 0.f, 1.f, 0, pre);
        add_v<<<dim3(HDN / 256, BB), 256, 0, stream>>>(t, query_pos, vbuf, Wv, pre);
        // DEQ fixed point: 15 damped iterations
        for (int it = 0; it < DEQ_ITERS; ++it) {
            pack(z, HID, zp);
            gemm(zp, HID, Wz_p, HDN, nullptr, pre, nullptr, 0.f, 1.f, 1, hdn);   // hdn = tanh(z@Wz + pre)
            pack(hdn, HDN, dp);
            gemm(dp, HDN, Wo_p, HID, bo, nullptr, z, 1.f - RELAX, RELAX, 0, z);  // z = 0.5z + 0.5(hdn@Wo+bo)
        }
        // liquid cell
        pack(z, HID, zp);
        pack(h, HID, hp);
        gemm(zp, HID, Wg_p, HID, bg, nullptr, nullptr, 0.f, 1.f, 0, tg);
        gemm(hp, HID, Ug_p, HID, nullptr, tg, nullptr, 0.f, 1.f, 2, gbuf);       // g = sigmoid(zWg+hUg+bg)
        gemm(zp, HID, Wl_p, HID, bl, nullptr, nullptr, 0.f, 1.f, 0, tl);
        gemm(hp, HID, Ul_p, HID, nullptr, tl, nullptr, 0.f, 1.f, 1, lbuf);       // l = tanh(zWl+hUl+bl)
        h_update<<<(BB * HID) / 256, 256, 0, stream>>>(gbuf, lbuf, h);
        // v_pred = norm(concat(z,h) @ Wzv + bzv)
        concat_zh<<<(BB * HDN) / 256, 256, 0, stream>>>(z, h, zh);
        pack(zh, HDN, zhp);
        gemm(zhp, HDN, Wzv_p, MEMD, bzv, nullptr, nullptr, 0.f, 1.f, 0, vtmp);
        norm_rows<<<BB, 128, 0, stream>>>(vtmp, vpred, MEMD);
        // VSA write: decay written slots, then deterministic scatter-add
        mark_written<<<1, 256, 0, stream>>>(t, write_pos, topi, written);
        apply_decay<<<NSLOTS, 256, 0, stream>>>(written, memory);
        scatter_write<<<MEMD / 256, 256, 0, stream>>>(t, write_pos, topi, w_slot, kvec, vpred, memory);
        // scaled cosine logits captured at each sequence's query step
        logits_k<<<dim3(NVAL / 256, BB), 256, 0, stream>>>(t, query_pos, vpred, Vp, scale, out);
    }
}
